// GraphSAGEBackbone_16982300688537
// MI455X (gfx1250) — compile-verified, matrix operand order checked by reading, not run on hardware
//
#include <hip/hip_runtime.h>

// GraphSAGE backbone for MI455X (gfx1250, wave32).
//
// Roofline: edge scatter (~1.7GB traffic/layer @ 23.3TB/s) dominates; the
// dense part (6.55 GFLOP/layer) is fused into a single [agg|x] (Nx256) @
// [Wl;Wr] (256x128) WMMA GEMM with f16 inputs / f32 accumulation, with bias,
// row L2-normalize and ReLU fused in the epilogue so output memory is touched
// exactly once.

#define D_FEAT 128
#define K_CAT  256

typedef __attribute__((ext_vector_type(16))) _Float16 v16h;
typedef __attribute__((ext_vector_type(8)))  float    v8f;

// ---------------------------------------------------------------- zeroing ---
__global__ void zero_kernel(float4* __restrict__ p, int n4) {
  int i = blockIdx.x * blockDim.x + threadIdx.x;
  if (i < n4) {
    float4 z = {0.f, 0.f, 0.f, 0.f};
    p[i] = z;
  }
}

// ------------------------------------------------- weight pack (once/layer) ---
// Bt[j][k] (f16, K contiguous) with k<128 -> Wl[k][j], k>=128 -> Wr[k-128][j].
// Transposed layout makes each WMMA B-fragment K-pair a single dword load.
__global__ void pack_weights(const float* __restrict__ Wl,
                             const float* __restrict__ Wr,
                             _Float16* __restrict__ Bt) {
  int i = blockIdx.x * blockDim.x + threadIdx.x;   // 0 .. 128*256-1
  if (i >= D_FEAT * K_CAT) return;
  int j = i >> 8;        // output column 0..127
  int k = i & 255;       // concat-K 0..255
  float v = (k < D_FEAT) ? Wl[k * D_FEAT + j] : Wr[(k - D_FEAT) * D_FEAT + j];
  Bt[(size_t)j * K_CAT + k] = (_Float16)v;
}

// -------------------------------------------------------- edge scatter-add ---
// One wave (32 lanes) per edge: lane handles a float4 (4 features) of x[src],
// atomically accumulated into agg[dst]; lane 0 bumps the degree counter.
__global__ void edge_scatter(const float* __restrict__ x,
                             const int* __restrict__ src,
                             const int* __restrict__ dst,
                             float* __restrict__ agg,
                             float* __restrict__ cnt, int E) {
  long long tid = (long long)blockIdx.x * blockDim.x + threadIdx.x;
  int e    = (int)(tid >> 5);
  int lane = (int)(tid & 31);
  if (e >= E) return;
  int s = src[e];
  int d = dst[e];
  float4 v = ((const float4*)(x + (long long)s * D_FEAT))[lane];
  float* ap = agg + (long long)d * D_FEAT + lane * 4;
  atomicAdd(ap + 0, v.x);
  atomicAdd(ap + 1, v.y);
  atomicAdd(ap + 2, v.z);
  atomicAdd(ap + 3, v.w);
  if (lane == 0) atomicAdd(cnt + d, 1.0f);
}

// ------------------------------------- fused mean + WMMA GEMM + norm + relu ---
// Block = 128 threads = 4 waves, computes 16 output rows:
//   A tile [16][256] f16 in LDS = [agg/max(cnt,1) | x]
//   each wave owns 2 of the 8 16-col tiles; K=256 -> 8 wmma_f32_16x16x32_f16
//   epilogue: +bias -> LDS, row L2 norm, relu, single global store.
__global__ void __launch_bounds__(128)
sage_gemm(const float* __restrict__ agg, const float* __restrict__ cnt,
          const float* __restrict__ xin, const _Float16* __restrict__ Bt,
          const float* __restrict__ bias, float* __restrict__ out, int n) {
  __shared__ __align__(16) _Float16 Alds[16][K_CAT];   // 8 KB
  __shared__ __align__(16) float    Olds[16][D_FEAT];  // 8 KB
  __shared__ float inv_norm[16];

  const int t = threadIdx.x;
  const long long rowg0 = (long long)blockIdx.x * 16;

  // ---- stage A tile: 8 threads per row, 16 cols each (f32 -> f16) ----
  {
    int row = t >> 3;                 // 0..15
    int cb  = (t & 7) * 16;           // 0,16,..,112
    long long r = rowg0 + row;
    if (r >= n) r = n - 1;            // clamp (safe duplicate reads)
    float inv = 1.0f / fmaxf(cnt[r], 1.0f);
    const float4* ap = (const float4*)(agg + r * D_FEAT);
    const float4* xp = (const float4*)(xin + r * D_FEAT);
#pragma unroll
    for (int q = 0; q < 4; ++q) {
      float4 av = ap[(cb >> 2) + q];
      float4 xv = xp[(cb >> 2) + q];
      int c0 = cb + q * 4;
      Alds[row][c0 + 0] = (_Float16)(av.x * inv);
      Alds[row][c0 + 1] = (_Float16)(av.y * inv);
      Alds[row][c0 + 2] = (_Float16)(av.z * inv);
      Alds[row][c0 + 3] = (_Float16)(av.w * inv);
      Alds[row][D_FEAT + c0 + 0] = (_Float16)xv.x;
      Alds[row][D_FEAT + c0 + 1] = (_Float16)xv.y;
      Alds[row][D_FEAT + c0 + 2] = (_Float16)xv.z;
      Alds[row][D_FEAT + c0 + 3] = (_Float16)xv.w;
    }
  }
  __syncthreads();

  // ---- WMMA main loop ----
  // ISA 16-bit operand layout: lane%16 selects row (A) / col (B); lane/16
  // selects K-half; each dword holds K=(k,k+1):
  //   pair p<4 : k = 8*g + 2p ;  pair p>=4 : k = 16 + 8*g + (2p-8)
  const int lane = t & 31;
  const int wid  = t >> 5;          // 0..3
  const int g    = lane >> 4;       // K-half select
  const int c16  = lane & 15;       // row (A) / col (B)
  const unsigned* __restrict__ Btu = (const unsigned*)Bt;
  const unsigned* __restrict__ arow = (const unsigned*)&Alds[c16][0];

#pragma unroll
  for (int jj = 0; jj < 2; ++jj) {
    const int jbase = (wid * 2 + jj) * 16;
    const long long bbase = ((long long)(jbase + c16) * K_CAT) >> 1; // dwords
    v8f acc = {};
#pragma unroll
    for (int kt = 0; kt < 8; ++kt) {
      const int kbase = kt * 32;
      union { unsigned u[8]; v16h v; } af, bf;
#pragma unroll
      for (int p = 0; p < 8; ++p) {
        int kk = (p < 4) ? (8 * g + 2 * p) : (16 + 8 * g + (2 * p - 8));
        af.u[p] = arow[(kbase + kk) >> 1];
        bf.u[p] = Btu[bbase + ((kbase + kk) >> 1)];
      }
      acc = __builtin_amdgcn_wmma_f32_16x16x32_f16(
          /*neg_a=*/false, af.v, /*neg_b=*/false, bf.v,
          /*c_mod=*/(short)0, acc, /*reuse_a=*/false, /*reuse_b=*/false);
    }
    float b = bias[jbase + c16];
#pragma unroll
    for (int v = 0; v < 8; ++v) {   // C/D layout: VGPR v -> M = v + 8*g
      Olds[v + 8 * g][jbase + c16] = acc[v] + b;
    }
  }
  __syncthreads();

  // ---- row L2 norms ----
  if (t < 16) {
    float ss = 0.f;
#pragma unroll 4
    for (int j = 0; j < D_FEAT; ++j) {
      float v = Olds[t][j];
      ss += v * v;
    }
    inv_norm[t] = 1.0f / fmaxf(sqrtf(ss), 1e-12f);
  }
  __syncthreads();

  // ---- normalize + relu + store ----
  {
    int row = t >> 3;
    int cb  = (t & 7) * 16;
    long long r = rowg0 + row;
    if (r < n) {
      float inv = inv_norm[row];
      float* op = out + r * D_FEAT;
#pragma unroll
      for (int c = cb; c < cb + 16; ++c) {
        op[c] = fmaxf(Olds[row][c] * inv, 0.f);
      }
    }
  }
}

// ------------------------------------------------------------------- launch ---
extern "C" void kernel_launch(void* const* d_in, const int* in_sizes, int n_in,
                              void* d_out, int out_size, void* d_ws, size_t ws_size,
                              hipStream_t stream) {
  const float* x   = (const float*)d_in[0];
  const int*   ei  = (const int*)d_in[1];     // [2, E] int32
  const float* Wl1 = (const float*)d_in[2];
  const float* bl1 = (const float*)d_in[3];
  const float* Wr1 = (const float*)d_in[4];
  const float* Wl2 = (const float*)d_in[5];
  const float* bl2 = (const float*)d_in[6];
  const float* Wr2 = (const float*)d_in[7];

  const int n = in_sizes[0] / D_FEAT;   // 100000 nodes
  const int E = in_sizes[1] / 2;        // 1600000 edges
  const int* src = ei;
  const int* dst = ei + E;

  // workspace layout: agg | cnt | H1 | Bt1 | Bt2  (~103 MB)
  char* ws = (char*)d_ws;
  float* aggp = (float*)ws;      ws += (size_t)n * D_FEAT * sizeof(float);
  float* cntp = (float*)ws;      ws += (size_t)n * sizeof(float);
  float* H1   = (float*)ws;      ws += (size_t)n * D_FEAT * sizeof(float);
  _Float16* Bt1 = (_Float16*)ws; ws += (size_t)D_FEAT * K_CAT * sizeof(_Float16);
  _Float16* Bt2 = (_Float16*)ws;

  const int wthreads = D_FEAT * K_CAT;                 // 32768
  pack_weights<<<(wthreads + 255) / 256, 256, 0, stream>>>(Wl1, Wr1, Bt1);
  pack_weights<<<(wthreads + 255) / 256, 256, 0, stream>>>(Wl2, Wr2, Bt2);

  const int zeroN4 = (n * (D_FEAT + 1) + 3) / 4;       // agg + cnt contiguous
  const long long ethreads = (long long)E * 32;
  const int eblocks = (int)((ethreads + 255) / 256);
  const int gblocks = (n + 15) / 16;

  // ---- layer 1 ----
  zero_kernel<<<(zeroN4 + 255) / 256, 256, 0, stream>>>((float4*)aggp, zeroN4);
  edge_scatter<<<eblocks, 256, 0, stream>>>(x, src, dst, aggp, cntp, E);
  sage_gemm<<<gblocks, 128, 0, stream>>>(aggp, cntp, x, Bt1, bl1, H1, n);

  // ---- layer 2 ----
  zero_kernel<<<(zeroN4 + 255) / 256, 256, 0, stream>>>((float4*)aggp, zeroN4);
  edge_scatter<<<eblocks, 256, 0, stream>>>(H1, src, dst, aggp, cntp, E);
  sage_gemm<<<gblocks, 128, 0, stream>>>(aggp, cntp, H1, Bt2, bl2, (float*)d_out, n);
}
